// MultiHeadAttention_30820685316572
// MI455X (gfx1250) — matrix operand order repaired and checked by reading, not compiled, not run on hardware
//
#include <hip/hip_runtime.h>
#include <hip/hip_bf16.h>
#include <cstdint>

// ---------------------------------------------------------------------------
// MI455X (gfx1250) fused multi-head attention, bf16 WMMA pipeline.
// wave32; matrix math via v_wmma_f32_16x16x32_bf16 (f32 accumulate);
// tile staging via global_load_async_to_lds_b128 (ASYNCcnt) when available,
// double-buffered LDS software pipeline either way.
// ---------------------------------------------------------------------------

typedef __attribute__((ext_vector_type(16))) __bf16 v16bf;
typedef __attribute__((ext_vector_type(8)))  __bf16 v8bf;
typedef __attribute__((ext_vector_type(8)))  float  v8f;
typedef int i32x4 __attribute__((vector_size(16)));   // matches builtin's V4i

#define D_EMB 1024
#define NHEAD 16
#define DK    64
#define BATCH 2
#define SEQ   2048
#define NTOK  (BATCH * SEQ)   // 4096 rows in every GEMM

static __device__ __forceinline__ v8f wmma_bf16(v16bf a, v16bf b, v8f c) {
  // 8 args: (neg_a, A, neg_b, B, c_mod, C, reuse_a, reuse_b)
  return __builtin_amdgcn_wmma_f32_16x16x32_bf16(false, a, false, b,
                                                 (short)0, c, false, false);
}

// ---------------------------------------------------------------------------
// Async global -> LDS staging (CDNA5 ASYNCcnt path) with synchronous fallback.
// Generic LDS pointers carry the LDS byte offset in their low 32 bits
// (aperture rule), so we synthesize address-space pointers via integer casts.
// Builtin signature: (AS1 i32x4* src, AS3 i32x4* dst, imm offset, imm cpol).
// ---------------------------------------------------------------------------
#if defined(__gfx1250__) &&                                              \
    __has_builtin(__builtin_amdgcn_global_load_async_to_lds_b128) &&     \
    __has_builtin(__builtin_amdgcn_s_wait_asynccnt)
#define USE_ASYNC 1
#else
#define USE_ASYNC 0
#endif

static __device__ __forceinline__ void async_cp16(const __bf16* g, __bf16* l) {
#if USE_ASYNC
  auto gp = (__attribute__((address_space(1))) i32x4*)(uintptr_t)g;
  auto lp = (__attribute__((address_space(3))) i32x4*)(uintptr_t)(
      unsigned)(uintptr_t)l;
  __builtin_amdgcn_global_load_async_to_lds_b128(gp, lp, 0, 0);
#else
  *reinterpret_cast<v8bf*>(l) = *reinterpret_cast<const v8bf*>(g);
#endif
}

static __device__ __forceinline__ void async_wait() {
#if USE_ASYNC
  __builtin_amdgcn_s_wait_asynccnt(0);
#endif
}

// ---------------------------------------------------------------------------
// fp32 -> bf16 conversion (grid-stride)
// ---------------------------------------------------------------------------
__global__ void cvt_f32_bf16(const float* __restrict__ src,
                             __bf16* __restrict__ dst, int n) {
  int i = blockIdx.x * blockDim.x + threadIdx.x;
  int stride = gridDim.x * blockDim.x;
  for (; i < n; i += stride) dst[i] = (__bf16)src[i];
}

// ---------------------------------------------------------------------------
// NT GEMM: C[n, m] = sum_k A[n,k] * W[m,k] + bias[m]
//   A: [4096, 1024] bf16 row-major, W: [1024, 1024] bf16 row-major.
// Block tile 128x64, 8 waves, wave tile 32x32 = 4 WMMA accumulators.
// Double-buffered LDS; A tile staged async, W tile transposed through VGPRs.
// mode 0: bf16 [b,h,s,d]; mode 1: bf16 [b,h,d,s]; mode 2: fp32 [n,m].
// ---------------------------------------------------------------------------
__global__ __launch_bounds__(256)
void gemm_nt(const __bf16* __restrict__ A, const __bf16* __restrict__ W,
             const float* __restrict__ bias, void* __restrict__ out, int mode) {
  __shared__ __align__(64) __bf16 lA[2][128][40];   // 20 KB
  __shared__ __align__(64) __bf16 lBt[2][32][80];   // 10 KB, [k][n]

  const int tid  = threadIdx.x;
  const int lane = tid & 31;
  const int wv   = tid >> 5;      // 0..7
  const int wr   = wv >> 1;       // 0..3
  const int wc   = wv & 1;        // 0..1
  const int rowBase = blockIdx.y * 128;
  const int colBase = blockIdx.x * 64;

  const v8f vzero = {0.f, 0.f, 0.f, 0.f, 0.f, 0.f, 0.f, 0.f};
  v8f acc[2][2];
  for (int i = 0; i < 2; ++i)
    for (int j = 0; j < 2; ++j) acc[i][j] = vzero;

  // A tile: 128 rows x 32 k = 512 chunks of 8 bf16; 2 chunks/thread, async.
  auto stageA = [&](int k0, int buf) {
#pragma unroll
    for (int c0 = 0; c0 < 2; ++c0) {
      int c = tid + c0 * 256;
      int row = c >> 2, seg = (c & 3) * 8;
      async_cp16(A + (size_t)(rowBase + row) * D_EMB + k0 + seg,
                 &lA[buf][row][seg]);
    }
  };
  // W tile: 64 n-rows x 32 k, loaded k-contiguous, stored transposed [k][n].
  v8bf wreg;
  auto loadW = [&](int k0) {
    int nl = tid >> 2, seg = (tid & 3) * 8;
    wreg = *reinterpret_cast<const v8bf*>(
        W + (size_t)(colBase + nl) * D_EMB + k0 + seg);
  };
  auto storeW = [&](int buf) {
    int nl = tid >> 2, seg = (tid & 3) * 8;
#pragma unroll
    for (int u = 0; u < 8; ++u) lBt[buf][seg + u][nl] = wreg[u];
  };

  stageA(0, 0);
  loadW(0);
  storeW(0);
  async_wait();
  __syncthreads();

  int cur = 0;
  for (int k0 = 0; k0 < D_EMB; k0 += 32) {
    const int nxt = cur ^ 1;
    const bool more = (k0 + 32) < D_EMB;
    if (more) { stageA(k0 + 32, nxt); loadW(k0 + 32); }   // in flight

    // ---- fragments (ISA VGPR layouts) ----
    // A 16x32: M = lane&15, K = h + 8*((lane>>4) + (h>>3))
    v16bf af[2];
#pragma unroll
    for (int i = 0; i < 2; ++i) {
      int m  = wr * 32 + i * 16 + (lane & 15);
      int lo = (lane >> 4) * 8;
      v8bf l = *reinterpret_cast<const v8bf*>(&lA[cur][m][lo]);
      v8bf h = *reinterpret_cast<const v8bf*>(&lA[cur][m][lo + 16]);
#pragma unroll
      for (int u = 0; u < 8; ++u) { af[i][u] = l[u]; af[i][u + 8] = h[u]; }
    }
    // B 32x16: N = lane&15, K = s + 8*((lane>>4) + (s>>3)) (same as A)
    v16bf bfr[2];
#pragma unroll
    for (int j = 0; j < 2; ++j) {
      int n0 = wc * 32 + j * 16;
#pragma unroll
      for (int s = 0; s < 16; ++s)
        bfr[j][s] = lBt[cur][(s & 7) + 8 * (lane >> 4) + 16 * (s >> 3)][n0 + (lane & 15)];
    }

#pragma unroll
    for (int i = 0; i < 2; ++i)
#pragma unroll
      for (int j = 0; j < 2; ++j)
        acc[i][j] = wmma_bf16(af[i], bfr[j], acc[i][j]);

    if (more) storeW(nxt);
    async_wait();
    __syncthreads();
    cur = nxt;
  }

  // ---- epilogue: C layout lane -> (M = r + 8*(lane>>4), N = lane&15) ----
#pragma unroll
  for (int i = 0; i < 2; ++i) {
#pragma unroll
    for (int j = 0; j < 2; ++j) {
#pragma unroll
      for (int r = 0; r < 8; ++r) {
        int m = rowBase + wr * 32 + i * 16 + r + 8 * (lane >> 4);
        int c = colBase + wc * 32 + j * 16 + (lane & 15);
        float v = acc[i][j][r] + bias[c];
        int b = m >> 11, s = m & (SEQ - 1);
        int h = c >> 6,  d = c & (DK - 1);
        if (mode == 0) {
          ((__bf16*)out)[((size_t)((b * NHEAD + h) * SEQ) + s) * DK + d] =
              (__bf16)v;
        } else if (mode == 1) {
          ((__bf16*)out)[((size_t)((b * NHEAD + h) * DK) + d) * SEQ + s] =
              (__bf16)v;
        } else {
          ((float*)out)[(size_t)m * D_EMB + c] = v;
        }
      }
    }
  }
}

// ---------------------------------------------------------------------------
// Flash attention (non-causal, online softmax).
//   q : [B*H, S, 64] bf16, kt: [B*H, 64, S] bf16, v: [B*H, S, 64] bf16
//   out: attn reshaped [B, S, 1024] bf16 (feeds O-projection GEMM)
// Block = 4 waves x 16 query rows; K/V tiles (32 keys) staged ONCE per block
// into double-buffered LDS (async), shared by all 4 waves.
// Per tile per wave: 4 WMMAs (scores, K=64) + 4 WMMAs (P.V, K=32).
// ---------------------------------------------------------------------------
__global__ __launch_bounds__(128)
void flash_attn(const __bf16* __restrict__ qg, const __bf16* __restrict__ ktg,
                const __bf16* __restrict__ vg, __bf16* __restrict__ outg) {
  __shared__ __align__(64) __bf16 ktl[2][64][48];   // 12 KB, [d][s-tile]
  __shared__ __align__(64) __bf16 vl[2][32][80];    // 10 KB, [s-tile][d]
  __shared__ __align__(64) __bf16 pl[4][16][40];    //  5 KB, per-wave P

  const int tid  = threadIdx.x;
  const int lane = tid & 31;
  const int w    = tid >> 5;                       // wave 0..3
  const int bh   = blockIdx.y;                     // 0..31
  const int qBase = blockIdx.x * 64 + w * 16;      // global query row base

  const __bf16* qhead  = qg  + (size_t)bh * SEQ * DK;
  const __bf16* kthead = ktg + (size_t)bh * DK * SEQ;
  const __bf16* vhead  = vg  + (size_t)bh * SEQ * DK;

  // Stage one 32-key tile of K^T (64x32) and V (32x64): 2x256 16B chunks.
  auto stageKV = [&](int jB, int buf) {
#pragma unroll
    for (int c0 = 0; c0 < 2; ++c0) {
      int c = tid + c0 * 128;
      int d = c >> 2, seg = (c & 3) * 8;
      async_cp16(kthead + (size_t)d * SEQ + jB + seg, &ktl[buf][d][seg]);
    }
#pragma unroll
    for (int c0 = 0; c0 < 2; ++c0) {
      int c = tid + c0 * 128;
      int j = c >> 3, seg = (c & 7) * 8;
      async_cp16(vhead + (size_t)(jB + j) * DK + seg, &vl[buf][j][seg]);
    }
  };

  // Q fragments (A layout), resident for the whole kernel.
  v16bf qf[2];
  {
    const __bf16* qrow = qhead + (size_t)(qBase + (lane & 15)) * DK;
    int lo = (lane >> 4) * 8;
#pragma unroll
    for (int kf = 0; kf < 2; ++kf) {
      v8bf l = *reinterpret_cast<const v8bf*>(qrow + kf * 32 + lo);
      v8bf h = *reinterpret_cast<const v8bf*>(qrow + kf * 32 + lo + 16);
#pragma unroll
      for (int u = 0; u < 8; ++u) { qf[kf][u] = l[u]; qf[kf][u + 8] = h[u]; }
    }
  }

  const v8f vzero = {0.f, 0.f, 0.f, 0.f, 0.f, 0.f, 0.f, 0.f};
  v8f o[4];
#pragma unroll
  for (int t = 0; t < 4; ++t) o[t] = vzero;
  float mrow[8], lrow[8];
#pragma unroll
  for (int r = 0; r < 8; ++r) { mrow[r] = -1e30f; lrow[r] = 0.f; }

  stageKV(0, 0);
  async_wait();
  __syncthreads();

  int cur = 0;
  for (int jB = 0; jB < SEQ; jB += 32) {
    const int nxt = cur ^ 1;
    const bool more = (jB + 32) < SEQ;
    if (more) stageKV(jB + 32, nxt);   // async, in flight during compute

    // ---- scores: S(16x32) = Q(16x64) . K^T(64x32) from LDS ----
    v8f s0 = vzero, s1 = vzero;
#pragma unroll
    for (int kf = 0; kf < 2; ++kf) {
      v16bf b0, b1;
#pragma unroll
      for (int s = 0; s < 16; ++s) {
        int kk = kf * 32 + (s & 7) + 8 * (lane >> 4) + 16 * (s >> 3);
        b0[s] = ktl[cur][kk][lane & 15];
        b1[s] = ktl[cur][kk][16 + (lane & 15)];
      }
      s0 = wmma_bf16(qf[kf], b0, s0);
      s1 = wmma_bf16(qf[kf], b1, s1);
    }

    // ---- online softmax (row stats via shfl-xor within 16-lane halves) ----
#pragma unroll
    for (int r = 0; r < 8; ++r) {
      float a0 = s0[r] * 0.125f;   // 1/sqrt(64)
      float a1 = s1[r] * 0.125f;
      float mx = fmaxf(a0, a1);
#pragma unroll
      for (int off = 1; off < 16; off <<= 1)
        mx = fmaxf(mx, __shfl_xor(mx, off, 32));
      float mnew = fmaxf(mrow[r], mx);
      float sc   = __expf(mrow[r] - mnew);
      float e0   = __expf(a0 - mnew);
      float e1   = __expf(a1 - mnew);
      float sum  = e0 + e1;
#pragma unroll
      for (int off = 1; off < 16; off <<= 1)
        sum += __shfl_xor(sum, off, 32);
      lrow[r] = lrow[r] * sc + sum;
      mrow[r] = mnew;
      int ml = r + 8 * (lane >> 4);
      pl[w][ml][lane & 15]        = (__bf16)e0;
      pl[w][ml][16 + (lane & 15)] = (__bf16)e1;
      o[0][r] *= sc; o[1][r] *= sc; o[2][r] *= sc; o[3][r] *= sc;
    }
    __syncthreads();

    // ---- P (C layout -> A layout via LDS), then O += P . V ----
    v16bf pf;
    {
      int m  = lane & 15;
      int lo = (lane >> 4) * 8;
      v8bf l = *reinterpret_cast<const v8bf*>(&pl[w][m][lo]);
      v8bf h = *reinterpret_cast<const v8bf*>(&pl[w][m][lo + 16]);
#pragma unroll
      for (int u = 0; u < 8; ++u) { pf[u] = l[u]; pf[u + 8] = h[u]; }
    }
#pragma unroll
    for (int t = 0; t < 4; ++t) {
      v16bf vf;
#pragma unroll
      for (int s = 0; s < 16; ++s)
        vf[s] = vl[cur][(s & 7) + 8 * (lane >> 4) + 16 * (s >> 3)][t * 16 + (lane & 15)];
      o[t] = wmma_bf16(pf, vf, o[t]);
    }

    async_wait();
    __syncthreads();
    cur = nxt;
  }

  // ---- finalize: O /= l, scatter to [B, S, H*64] bf16 ----
  const int b = bh >> 4, h = bh & 15;
#pragma unroll
  for (int r = 0; r < 8; ++r) {
    float inv = 1.0f / lrow[r];
    int sg = qBase + r + 8 * (lane >> 4);
    size_t nrow = ((size_t)b * SEQ + sg) * D_EMB + h * DK;
#pragma unroll
    for (int t = 0; t < 4; ++t)
      outg[nrow + t * 16 + (lane & 15)] = (__bf16)(o[t][r] * inv);
  }
}

// ---------------------------------------------------------------------------
// Host-side orchestration. Workspace need: 32M bf16 elements = 64 MB.
// ---------------------------------------------------------------------------
extern "C" void kernel_launch(void* const* d_in, const int* in_sizes, int n_in,
                              void* d_out, int out_size, void* d_ws,
                              size_t ws_size, hipStream_t stream) {
  (void)in_sizes; (void)n_in; (void)out_size; (void)ws_size;

  const float* Q  = (const float*)d_in[0];
  const float* K  = (const float*)d_in[1];
  const float* V  = (const float*)d_in[2];
  const float* WQ = (const float*)d_in[3];
  const float* bQ = (const float*)d_in[4];
  const float* WK = (const float*)d_in[5];
  const float* bK = (const float*)d_in[6];
  const float* WV = (const float*)d_in[7];
  const float* bV = (const float*)d_in[8];
  const float* WO = (const float*)d_in[9];
  const float* bO = (const float*)d_in[10];

  const size_t NACT = (size_t)NTOK * D_EMB;    // 4M elems
  const size_t NW   = (size_t)D_EMB * D_EMB;   // 1M elems

  __bf16* p    = (__bf16*)d_ws;
  __bf16* Qb   = p; p += NACT;
  __bf16* Kb   = p; p += NACT;
  __bf16* Vb   = p; p += NACT;
  __bf16* WQb  = p; p += NW;
  __bf16* WKb  = p; p += NW;
  __bf16* WVb  = p; p += NW;
  __bf16* WOb  = p; p += NW;
  __bf16* qh   = p; p += NACT;   // [B,H,S,64]
  __bf16* kth  = p; p += NACT;   // [B,H,64,S]
  __bf16* vh   = p; p += NACT;   // [B,H,S,64]
  __bf16* attn = p; p += NACT;   // [B,S,1024]

  cvt_f32_bf16<<<2048, 256, 0, stream>>>(Q, Qb, (int)NACT);
  cvt_f32_bf16<<<2048, 256, 0, stream>>>(K, Kb, (int)NACT);
  cvt_f32_bf16<<<2048, 256, 0, stream>>>(V, Vb, (int)NACT);
  cvt_f32_bf16<<<512, 256, 0, stream>>>(WQ, WQb, (int)NW);
  cvt_f32_bf16<<<512, 256, 0, stream>>>(WK, WKb, (int)NW);
  cvt_f32_bf16<<<512, 256, 0, stream>>>(WV, WVb, (int)NW);
  cvt_f32_bf16<<<512, 256, 0, stream>>>(WO, WOb, (int)NW);

  dim3 gg(D_EMB / 64, NTOK / 128);   // 16 x 32
  gemm_nt<<<gg, 256, 0, stream>>>(Qb, WQb, bQ, qh, 0);
  gemm_nt<<<gg, 256, 0, stream>>>(Kb, WKb, bK, kth, 1);
  gemm_nt<<<gg, 256, 0, stream>>>(Vb, WVb, bV, vh, 0);

  flash_attn<<<dim3(SEQ / 64, BATCH * NHEAD), 128, 0, stream>>>(qh, kth, vh,
                                                                attn);

  gemm_nt<<<gg, 256, 0, stream>>>(attn, WOb, bO, d_out, 2);
}
